// TensorizedParallelFFN_79250736546756
// MI455X (gfx1250) — matrix operand order repaired and checked by reading, not compiled, not run on hardware
//
#include <hip/hip_runtime.h>

// ---------------------------------------------------------------------------
// TensorizedParallelFFN (MoE-style dense FFN with LoRA) for MI455X / gfx1250.
//
// Math:  W_eff = W + 2*(lora_b @ lora_a)  (fold LoRA into weights, exact)
//        hidden = silu(X Wg_effT) * (X Wu_effT)          [8192 x 8192]
//        out    = hidden @ Wd_eff_concatT                [8192 x 2048]
// Precision: bf16 operands, f32 WMMA accumulation (v_wmma_f32_16x16x32_bf16).
// Compute-bound: ~825 GFLOP vs ~0.3 GB minimum traffic (2.5 kFLOP/B).
// A tiles staged global->LDS via CDNA5 async-to-LDS path (ASYNCcnt); B weight
// fragments pre-swizzled to the WMMA register layout and WGP-scope prefetched.
// ---------------------------------------------------------------------------

typedef __attribute__((ext_vector_type(16))) __bf16 v16bf;
typedef __attribute__((ext_vector_type(8)))  float  v8f;
typedef __attribute__((ext_vector_type(4)))  int    v4i_ll;

struct U32x8 { uint4 a, b; };   // 32B container for one bf16x16 WMMA fragment

#define NTOK  8192   // B*S
#define HDIM  2048
#define KEXP  8
#define BLKW  1024
#define NFFN  8192   // KEXP*BLKW
#define RRANK 16

// ---- CDNA5 async global->LDS (guarded; falls back if builtin absent) ------
#if defined(__has_builtin)
#  if __has_builtin(__builtin_amdgcn_global_load_async_to_lds_b128) && \
      __has_builtin(__builtin_amdgcn_s_wait_asynccnt)
#    define USE_ASYNC_LDS 1
#  endif
#endif
#ifndef USE_ASYNC_LDS
#  define USE_ASYNC_LDS 0
#endif

#if USE_ASYNC_LDS
typedef __attribute__((address_space(1))) v4i_ll* as1_v4i_p;   // global int4*
typedef __attribute__((address_space(3))) v4i_ll* as3_v4i_p;   // LDS int4*
#endif

__device__ __forceinline__ void stage_a_32B(const __bf16* gsrc, char* ldst)
{
#if USE_ASYNC_LDS
    __builtin_amdgcn_global_load_async_to_lds_b128(
        (as1_v4i_p)(gsrc),     (as3_v4i_p)(ldst),      0, 0);
    __builtin_amdgcn_global_load_async_to_lds_b128(
        (as1_v4i_p)(gsrc + 8), (as3_v4i_p)(ldst + 16), 0, 0);
#else
    uint4 a0 = *(const uint4*)gsrc;
    uint4 a1 = *(const uint4*)(gsrc + 8);
    *(uint4*)ldst        = a0;
    *(uint4*)(ldst + 16) = a1;
#endif
}

__device__ __forceinline__ void stage_a_wait()
{
#if USE_ASYNC_LDS
    __builtin_amdgcn_s_wait_asynccnt(0);
#endif
}

// ---------------------------------------------------------------------------
// Weight prep: W_eff = W + 2*B@A, cast bf16, store in WMMA B-operand fragment
// layout: fragment(tn,tk) = 16(N)x32(K); elem addr = frag*512 + lane*16 + j,
// lane = (k%32<16 ? 0:16) + n%16, j = k%16  (lanes0-15 K0..15, lanes16-31 K16..31)
// ---------------------------------------------------------------------------
__global__ void prep_inproj(const float* __restrict__ W,
                            const float* __restrict__ la,   // [K,R,H]
                            const float* __restrict__ lb,   // [K,BLK,R]
                            __bf16* __restrict__ dst)       // B-frag, Kdim=H
{
    int idx = blockIdx.x * blockDim.x + threadIdx.x;
    if (idx >= NFFN * HDIM) return;
    int h = idx % HDIM, o = idx / HDIM;
    int ke = o >> 10;
    float v = W[(size_t)o * HDIM + h];
    const float* lbp = lb + (size_t)o * RRANK;
    const float* lap = la + (size_t)ke * RRANK * HDIM + h;
    float d = 0.f;
#pragma unroll
    for (int r = 0; r < RRANK; ++r) d += lbp[r] * lap[(size_t)r * HDIM];
    v += 2.0f * d;
    int tn = o >> 4, tk = h >> 5, kk = h & 31;
    int lane = ((kk & 16) ? 16 : 0) + (o & 15);
    dst[((size_t)(tn * (HDIM / 32) + tk) << 9) + lane * 16 + (kk & 15)] = (__bf16)v;
}

__global__ void prep_down(const float* __restrict__ W,     // [K,H,BLK]
                          const float* __restrict__ la,    // [K,R,BLK]
                          const float* __restrict__ lb,    // [K,H,R]
                          __bf16* __restrict__ dst)        // B-frag, Kdim=NFFN
{
    int idx = blockIdx.x * blockDim.x + threadIdx.x;
    if (idx >= HDIM * NFFN) return;
    int i = idx % NFFN, o = idx / NFFN;          // i: concat inner dim, o: H out
    int ke = i >> 10, bi = i & (BLKW - 1);
    float v = W[(size_t)ke * HDIM * BLKW + (size_t)o * BLKW + bi];
    const float* lbp = lb + ((size_t)ke * HDIM + o) * RRANK;
    const float* lap = la + (size_t)ke * RRANK * BLKW + bi;
    float d = 0.f;
#pragma unroll
    for (int r = 0; r < RRANK; ++r) d += lbp[r] * lap[(size_t)r * BLKW];
    v += 2.0f * d;
    int tn = o >> 4, tk = i >> 5, kk = i & 31;
    int lane = ((kk & 16) ? 16 : 0) + (o & 15);
    dst[((size_t)(tn * (NFFN / 32) + tk) << 9) + lane * 16 + (kk & 15)] = (__bf16)v;
}

__global__ void prep_x(const float* __restrict__ x, __bf16* __restrict__ xb, int n)
{
    int idx = blockIdx.x * blockDim.x + threadIdx.x;
    if (idx < n) xb[idx] = (__bf16)x[idx];
}

// ---------------------------------------------------------------------------
// LDS A-tile staging: 128 rows x 32 bf16, row stride 80B (pad -> conflict-free
// ds_load_b128 across 16 lanes). Double buffered, one barrier per k-step.
// A-fragment register layout read as two 16B LDS loads per lane:
//   lanes 0-15 : bytes {0..15, 16..31}  (K 0..15)
//   lanes16-31 : bytes {32..47, 48..63} (K 16..31)
// ---------------------------------------------------------------------------
#define LDS_ROW 80
#define LDS_BUF (128 * LDS_ROW)

__device__ __forceinline__ v16bf lds_afrag(const char* buf, int row, int off0)
{
    const char* p = buf + row * LDS_ROW + off0;
    U32x8 t;
    t.a = *(const uint4*)p;
    t.b = *(const uint4*)(p + 16);
    return __builtin_bit_cast(v16bf, t);
}

__device__ __forceinline__ v16bf gbl_bfrag(const __bf16* base, size_t elemOff)
{
    U32x8 t;
    t.a = *(const uint4*)(base + elemOff);
    t.b = *(const uint4*)(base + elemOff + 8);
    return __builtin_bit_cast(v16bf, t);
}

// ---------------------------------------------------------------------------
// GEMM1 (fused gate+up+silu*mul): WG tile 128M x 128N, 8 waves (2x4),
// wave = 64M x 32N per matrix, dual accumulators, K = HDIM.
// ---------------------------------------------------------------------------
__global__ __launch_bounds__(256) void ffn_gemm1(
    const __bf16* __restrict__ X,    // [NTOK, HDIM] row-major
    const __bf16* __restrict__ Wg,   // B-frag
    const __bf16* __restrict__ Wu,   // B-frag
    __bf16* __restrict__ Hid)        // [NTOK, NFFN] row-major
{
    __shared__ __align__(16) char lds[2 * LDS_BUF];
    const int tid  = threadIdx.x;
    const int wid  = tid >> 5, lane = tid & 31;
    const int wm   = wid >> 2, wn = wid & 3;
    const int m0   = blockIdx.y * 128;
    const int n0   = blockIdx.x * 128;
    const int KT   = HDIM / 32;

    v8f accG[4][2], accU[4][2];
    const v8f vz = {0.f, 0.f, 0.f, 0.f, 0.f, 0.f, 0.f, 0.f};
#pragma unroll
    for (int i = 0; i < 4; ++i)
#pragma unroll
        for (int j = 0; j < 2; ++j) { accG[i][j] = vz; accU[i][j] = vz; }

    // cooperative A-tile staging: thread -> (row, 32B half)
    const int crow = tid >> 1, chalf = tid & 1;
    const __bf16* gA = X + (size_t)(m0 + crow) * HDIM + chalf * 16;
    const int ldsoff = crow * LDS_ROW + chalf * 32;

    stage_a_32B(gA, lds + ldsoff);     // preload k-step 0 into buffer 0
    stage_a_wait();
    __syncthreads();

    const int l16  = lane & 15;
    const int off0 = (lane & 16) ? 32 : 0;
    const int tnb  = (n0 >> 4) + wn * 2;

    for (int kt = 0; kt < KT; ++kt) {
        const char* buf = lds + (kt & 1) * LDS_BUF;
        if (kt + 1 < KT)   // async prefetch next A tile into other buffer
            stage_a_32B(gA + (size_t)(kt + 1) * 32,
                        lds + ((kt + 1) & 1) * LDS_BUF + ldsoff);

        v16bf afr[4];
#pragma unroll
        for (int i = 0; i < 4; ++i)
            afr[i] = lds_afrag(buf, wm * 64 + i * 16 + l16, off0);

        v16bf bg[2], bu[2];
#pragma unroll
        for (int j = 0; j < 2; ++j) {
            size_t fo = (((size_t)(tnb + j) * KT + kt) << 9) + lane * 16;
            bg[j] = gbl_bfrag(Wg, fo);
            bu[j] = gbl_bfrag(Wu, fo);
            if (kt + 4 < KT) {         // near-scope stream prefetch: +4KB
                __builtin_prefetch(Wg + fo + 4 * 512, 0, 3);
                __builtin_prefetch(Wu + fo + 4 * 512, 0, 3);
            }
        }
#pragma unroll
        for (int i = 0; i < 4; ++i)
#pragma unroll
            for (int j = 0; j < 2; ++j) {
                accG[i][j] = __builtin_amdgcn_wmma_f32_16x16x32_bf16(
                    false, afr[i], false, bg[j], (short)0, accG[i][j], false, false);
                accU[i][j] = __builtin_amdgcn_wmma_f32_16x16x32_bf16(
                    false, afr[i], false, bu[j], (short)0, accU[i][j], false, false);
            }
        stage_a_wait();                // next tile fully landed in LDS
        __syncthreads();
    }

    // epilogue: hidden = silu(gate) * up, store bf16 row-major
    const int hi8 = (lane & 16) ? 8 : 0;
#pragma unroll
    for (int i = 0; i < 4; ++i) {
        const int rbase = m0 + wm * 64 + i * 16 + hi8;
#pragma unroll
        for (int j = 0; j < 2; ++j) {
            const int c = n0 + wn * 32 + j * 16 + l16;
#pragma unroll
            for (int g = 0; g < 8; ++g) {
                float gg = accG[i][j][g];
                float uu = accU[i][j][g];
                float hv = (gg / (1.0f + __expf(-gg))) * uu;
                Hid[(size_t)(rbase + g) * NFFN + c] = (__bf16)hv;
            }
        }
    }
}

// ---------------------------------------------------------------------------
// GEMM2 (down, expert-sum via K-reduction): WG 128M x 256N, wave 64x64,
// K = NFFN = 8192, fp32 output to d_out.
// ---------------------------------------------------------------------------
__global__ __launch_bounds__(256) void ffn_gemm2(
    const __bf16* __restrict__ Hd,   // [NTOK, NFFN] row-major
    const __bf16* __restrict__ Wd,   // B-frag, Kdim=NFFN
    float* __restrict__ Out)         // [NTOK, HDIM]
{
    __shared__ __align__(16) char lds[2 * LDS_BUF];
    const int tid  = threadIdx.x;
    const int wid  = tid >> 5, lane = tid & 31;
    const int wm   = wid >> 2, wn = wid & 3;
    const int m0   = blockIdx.y * 128;
    const int n0   = blockIdx.x * 256;
    const int KT   = NFFN / 32;

    v8f acc[4][4];
    const v8f vz = {0.f, 0.f, 0.f, 0.f, 0.f, 0.f, 0.f, 0.f};
#pragma unroll
    for (int i = 0; i < 4; ++i)
#pragma unroll
        for (int j = 0; j < 4; ++j) acc[i][j] = vz;

    const int crow = tid >> 1, chalf = tid & 1;
    const __bf16* gA = Hd + (size_t)(m0 + crow) * NFFN + chalf * 16;
    const int ldsoff = crow * LDS_ROW + chalf * 32;

    stage_a_32B(gA, lds + ldsoff);
    stage_a_wait();
    __syncthreads();

    const int l16  = lane & 15;
    const int off0 = (lane & 16) ? 32 : 0;
    const int tnb  = (n0 >> 4) + wn * 4;

    for (int kt = 0; kt < KT; ++kt) {
        const char* buf = lds + (kt & 1) * LDS_BUF;
        if (kt + 1 < KT)
            stage_a_32B(gA + (size_t)(kt + 1) * 32,
                        lds + ((kt + 1) & 1) * LDS_BUF + ldsoff);

        v16bf afr[4];
#pragma unroll
        for (int i = 0; i < 4; ++i)
            afr[i] = lds_afrag(buf, wm * 64 + i * 16 + l16, off0);

        v16bf bfr[4];
#pragma unroll
        for (int j = 0; j < 4; ++j) {
            size_t fo = (((size_t)(tnb + j) * KT + kt) << 9) + lane * 16;
            bfr[j] = gbl_bfrag(Wd, fo);
            if (kt + 4 < KT) __builtin_prefetch(Wd + fo + 4 * 512, 0, 3);
        }

#pragma unroll
        for (int i = 0; i < 4; ++i)
#pragma unroll
            for (int j = 0; j < 4; ++j)
                acc[i][j] = __builtin_amdgcn_wmma_f32_16x16x32_bf16(
                    false, afr[i], false, bfr[j], (short)0, acc[i][j], false, false);
        stage_a_wait();
        __syncthreads();
    }

    const int hi8 = (lane & 16) ? 8 : 0;
#pragma unroll
    for (int i = 0; i < 4; ++i) {
        const int rbase = m0 + wm * 64 + i * 16 + hi8;
#pragma unroll
        for (int j = 0; j < 4; ++j) {
            const int c = n0 + wn * 64 + j * 16 + l16;
#pragma unroll
            for (int g = 0; g < 8; ++g)
                Out[(size_t)(rbase + g) * HDIM + c] = acc[i][j][g];
        }
    }
}

// ---------------------------------------------------------------------------
extern "C" void kernel_launch(void* const* d_in, const int* in_sizes, int n_in,
                              void* d_out, int out_size, void* d_ws, size_t ws_size,
                              hipStream_t stream)
{
    const float* x   = (const float*)d_in[0];
    const float* gw  = (const float*)d_in[1];
    const float* uw  = (const float*)d_in[2];
    const float* dw  = (const float*)d_in[3];
    const float* gla = (const float*)d_in[4];
    const float* glb = (const float*)d_in[5];
    const float* ula = (const float*)d_in[6];
    const float* ulb = (const float*)d_in[7];
    const float* dla = (const float*)d_in[8];
    const float* dlb = (const float*)d_in[9];
    float* out = (float*)d_out;

    char* ws = (char*)d_ws;
    const size_t SZ_X  = (size_t)NTOK * HDIM * 2;   // 33.5 MB bf16
    const size_t SZ_W  = (size_t)NFFN * HDIM * 2;   // 33.5 MB bf16 each
    __bf16* xb  = (__bf16*)(ws);
    __bf16* wgf = (__bf16*)(ws + SZ_X);
    __bf16* wuf = (__bf16*)(ws + SZ_X + SZ_W);
    __bf16* wdf = (__bf16*)(ws + SZ_X + 2 * SZ_W);
    __bf16* hid = (__bf16*)(ws + SZ_X + 3 * SZ_W);  // 134 MB bf16

    // phase 0: precision-cast x, fold LoRA into weights + swizzle to WMMA layout
    {
        const int nx = NTOK * HDIM;
        prep_x<<<dim3((nx + 255) / 256), dim3(256), 0, stream>>>(x, xb, nx);
        const int nw = NFFN * HDIM;
        prep_inproj<<<dim3((nw + 255) / 256), dim3(256), 0, stream>>>(gw, gla, glb, wgf);
        prep_inproj<<<dim3((nw + 255) / 256), dim3(256), 0, stream>>>(uw, ula, ulb, wuf);
        prep_down  <<<dim3((nw + 255) / 256), dim3(256), 0, stream>>>(dw, dla, dlb, wdf);
    }
    // phase 1: hidden = silu(x Wg^T) * (x Wu^T)   (fused, dual-accumulator)
    ffn_gemm1<<<dim3(NFFN / 128, NTOK / 128), dim3(256), 0, stream>>>(xb, wgf, wuf, hid);
    // phase 2: out = hidden @ Wd_concat^T   (expert sum == K reduction)
    ffn_gemm2<<<dim3(HDIM / 256, NTOK / 128), dim3(256), 0, stream>>>(hid, wdf, out);
}
